// Net_36369783063014
// MI455X (gfx1250) — compile-verified, matrix operand order checked by reading, not run on hardware
//
#include <hip/hip_runtime.h>

typedef __attribute__((ext_vector_type(16))) _Float16 v16h;
typedef __attribute__((ext_vector_type(8)))  float    v8f;
typedef __attribute__((ext_vector_type(4)))  float    v4f;

#define WGT  128      // 4 waves of 32
#define TSEQ 512
#define BT   16       // batch rows per workgroup

// ---------------- LDS layout (byte offsets) ----------------
#define OFF_LWHH0 0u        // 16 ntiles x 2 kfrags x 1024 B
#define OFF_LWIH1 32768u
#define OFF_LWHH1 65536u
#define OFF_GWIH0 98304u    // 12 x 2
#define OFF_GWHH0 122880u
#define OFF_GWIH1 147456u
#define OFF_GWHH1 172032u
// Hidden-state fragment buffers, double-buffered: 2 parities x 2 KB
#define OFF_H0    196608u
#define OFF_H1    200704u
#define OFF_HG0   204800u
#define OFF_HG1   208896u
// Input slice, transposed XT[t][m] (f32)
#define OFF_XT    212992u
// Biases / misc (f32)
#define OFF_LB0   245760u
#define OFF_LB1   246784u
#define OFF_GBI0  247808u
#define OFF_GBH0  248576u
#define OFF_GBI1  249344u
#define OFF_GBH1  250112u
#define OFF_WIH0  250880u
#define OFF_FCW   251904u
#define OFF_RED   252160u
#define LDS_TOTAL 256256u   // ~250 KB < 320 KB WGP LDS

extern __shared__ char smem[];

// Native CDNA5 transcendental tanh (TRANS32 pipe).
__device__ __forceinline__ float tanh_hw(float x) {
#if __has_builtin(__builtin_amdgcn_tanhf)
  return __builtin_amdgcn_tanhf(x);
#else
  float y;
  asm volatile("v_tanh_f32 %0, %1\n\tv_nop" : "=v"(y) : "v"(x));
  return y;
#endif
}
__device__ __forceinline__ float sig_hw(float x) {
  return __builtin_fmaf(0.5f, tanh_hw(0.5f * x), 0.5f);
}

// Swizzle a row-major [N][K] f32 weight into WMMA B-fragment layout (f16) in LDS.
__device__ __forceinline__ void swz_weight(const float* __restrict__ g, _Float16* dst,
                                           int N, int K, int tid) {
  const int kfr = K >> 5;
  for (int idx = tid; idx < N * K; idx += WGT) {
    int n = idx / K, k = idx - n * K;
    int kk = k & 31;
    int f = (n >> 4) * kfr + (k >> 5);
    int lane = (((kk >> 3) & 1) << 4) | (n & 15);
    int e = (kk & 7) + ((kk & 16) >> 1);
    dst[(f << 9) + (lane << 4) + e] = (_Float16)g[idx];
  }
}

__device__ __forceinline__ v16h ldsv(unsigned off) { return *(const v16h*)(smem + off); }

__global__ __launch_bounds__(WGT, 1) void rnn_fused(
    const float* __restrict__ x,
    const float* __restrict__ lw_ih0, const float* __restrict__ lw_hh0,
    const float* __restrict__ lb_ih0, const float* __restrict__ lb_hh0,
    const float* __restrict__ lw_ih1, const float* __restrict__ lw_hh1,
    const float* __restrict__ lb_ih1, const float* __restrict__ lb_hh1,
    const float* __restrict__ gw_ih0, const float* __restrict__ gw_hh0,
    const float* __restrict__ gb_ih0, const float* __restrict__ gb_hh0,
    const float* __restrict__ gw_ih1, const float* __restrict__ gw_hh1,
    const float* __restrict__ gb_ih1, const float* __restrict__ gb_hh1,
    const float* __restrict__ fc_w, const float* __restrict__ fc_b,
    float* __restrict__ out)
{
  const int tid  = threadIdx.x;
  const int lane = tid & 31;
  const int wv   = tid >> 5;
  const int wg   = blockIdx.x;

  float* LB0  = (float*)(smem + OFF_LB0);
  float* LB1  = (float*)(smem + OFF_LB1);
  float* GBI0 = (float*)(smem + OFF_GBI0);
  float* GBH0 = (float*)(smem + OFF_GBH0);
  float* GBI1 = (float*)(smem + OFF_GBI1);
  float* GBH1 = (float*)(smem + OFF_GBH1);
  float* WIH0 = (float*)(smem + OFF_WIH0);
  float* FCW  = (float*)(smem + OFF_FCW);
  float* RED  = (float*)(smem + OFF_RED);
  float* XT   = (float*)(smem + OFF_XT);

  // ---------------- one-time setup ----------------
  swz_weight(lw_hh0, (_Float16*)(smem + OFF_LWHH0), 256, 64, tid);
  swz_weight(lw_ih1, (_Float16*)(smem + OFF_LWIH1), 256, 64, tid);
  swz_weight(lw_hh1, (_Float16*)(smem + OFF_LWHH1), 256, 64, tid);
  swz_weight(gw_ih0, (_Float16*)(smem + OFF_GWIH0), 192, 64, tid);
  swz_weight(gw_hh0, (_Float16*)(smem + OFF_GWHH0), 192, 64, tid);
  swz_weight(gw_ih1, (_Float16*)(smem + OFF_GWIH1), 192, 64, tid);
  swz_weight(gw_hh1, (_Float16*)(smem + OFF_GWHH1), 192, 64, tid);
  for (int i = tid; i < 256; i += WGT) {
    LB0[i] = lb_ih0[i] + lb_hh0[i];
    LB1[i] = lb_ih1[i] + lb_hh1[i];
    WIH0[i] = lw_ih0[i];
  }
  for (int i = tid; i < 192; i += WGT) {
    GBI0[i] = gb_ih0[i]; GBH0[i] = gb_hh0[i];
    GBI1[i] = gb_ih1[i]; GBH1[i] = gb_hh1[i];
  }
  for (int i = tid; i < 64; i += WGT) FCW[i] = fc_w[i];
  {
    int* hz = (int*)(smem + OFF_H0);
    for (int i = tid; i < 4096; i += WGT) hz[i] = 0;
  }
  for (int idx = tid; idx < BT * TSEQ; idx += WGT) {
    int m = idx >> 9, t = idx & (TSEQ - 1);
    XT[t * BT + m] = x[(size_t)(wg * BT + m) * TSEQ + t];
  }
  __syncthreads();

  // ---------------- register-resident, loop-invariant data ----------------
  const int mrow = (lane >> 4) << 3;
  const int ncol = lane & 15;
  const int jcol = wv * 16 + ncol;
  const unsigned lb = (unsigned)lane << 5;

  v16h wL0[4][2], wL1I[4][2], wL1H[4][2];
  v16h wG0I[3][2], wG0H[3][2], wG1I[3][2], wG1H[3][2];
#pragma unroll
  for (int i = 0; i < 4; i++) {
    unsigned nt = (unsigned)(wv + 4 * i) * 2048u;
    wL0[i][0]  = ldsv(OFF_LWHH0 + nt + lb);        wL0[i][1]  = ldsv(OFF_LWHH0 + nt + 1024u + lb);
    wL1I[i][0] = ldsv(OFF_LWIH1 + nt + lb);        wL1I[i][1] = ldsv(OFF_LWIH1 + nt + 1024u + lb);
    wL1H[i][0] = ldsv(OFF_LWHH1 + nt + lb);        wL1H[i][1] = ldsv(OFF_LWHH1 + nt + 1024u + lb);
  }
#pragma unroll
  for (int i = 0; i < 3; i++) {
    unsigned nt = (unsigned)(wv + 4 * i) * 2048u;
    wG0I[i][0] = ldsv(OFF_GWIH0 + nt + lb);        wG0I[i][1] = ldsv(OFF_GWIH0 + nt + 1024u + lb);
    wG0H[i][0] = ldsv(OFF_GWHH0 + nt + lb);        wG0H[i][1] = ldsv(OFF_GWHH0 + nt + 1024u + lb);
    wG1I[i][0] = ldsv(OFF_GWIH1 + nt + lb);        wG1I[i][1] = ldsv(OFF_GWIH1 + nt + 1024u + lb);
    wG1H[i][0] = ldsv(OFF_GWHH1 + nt + lb);        wG1H[i][1] = ldsv(OFF_GWHH1 + nt + 1024u + lb);
  }
  float bL0[4], wn0[4], bL1[4], bGI0[3], bGH0[3], bGI1[3], bGH1[3];
#pragma unroll
  for (int i = 0; i < 4; i++) {
    int n = jcol + 64 * i;
    bL0[i] = LB0[n]; wn0[i] = WIH0[n]; bL1[i] = LB1[n];
  }
#pragma unroll
  for (int i = 0; i < 3; i++) {
    int n = jcol + 64 * i;
    bGI0[i] = GBI0[n]; bGH0[i] = GBH0[n]; bGI1[i] = GBI1[n]; bGH1[i] = GBH1[n];
  }
  const float fcwj = FCW[jcol];

  const unsigned hoff = ((unsigned)(jcol >> 5) << 10) + ((unsigned)((jcol >> 3) & 1) << 9)
                      + ((unsigned)((jcol & 7) + ((jcol & 16) >> 1)) << 1)
                      + ((unsigned)mrow << 5);

  float c0r[8], c1r[8], hg0r[8], hg1r[8];
#pragma unroll
  for (int r = 0; r < 8; r++) { c0r[r] = 0.f; c1r[r] = 0.f; hg0r[r] = 0.f; hg1r[r] = 0.f; }

  for (int t = 0; t < TSEQ; t++) {
    const unsigned rp = (unsigned)(t & 1) * 2048u;
    const unsigned wp = 2048u - rp;

    // ---- early hidden-path preloads (parity rp, writers >=1 barrier away) ----
    v16h l0a0 = ldsv(OFF_H0  + rp + lb), l0a1 = ldsv(OFF_H0  + rp + 1024u + lb);
    v16h l1h0 = ldsv(OFF_H1  + rp + lb), l1h1 = ldsv(OFF_H1  + rp + 1024u + lb);
    v16h g0h0 = ldsv(OFF_HG0 + rp + lb), g0h1 = ldsv(OFF_HG0 + rp + 1024u + lb);
    v4f x0 = *(const v4f*)(smem + OFF_XT + (unsigned)t * 64u + ((unsigned)mrow << 2));
    v4f x1 = *(const v4f*)(smem + OFF_XT + (unsigned)t * 64u + ((unsigned)mrow << 2) + 16u);

    // ================= LSTM layer 0 =================
    {
      v8f gx[4], gh[4];
#pragma unroll
      for (int i = 0; i < 4; i++) {
        v8f zz;
#pragma unroll
        for (int r = 0; r < 8; r++) zz[r] = 0.0f;
        gh[i] = __builtin_amdgcn_wmma_f32_16x16x32_f16(false, l0a1, false, wL0[i][1], (short)0, zz, false, false);
      }
#pragma unroll
      for (int i = 0; i < 4; i++) {
        v8f cx;
#pragma unroll
        for (int r = 0; r < 8; r++) cx[r] = (r < 4 ? x0[r] : x1[r - 4]) * wn0[i] + bL0[i];
        gx[i] = __builtin_amdgcn_wmma_f32_16x16x32_f16(false, l0a0, false, wL0[i][0], (short)0, cx, false, false);
      }
#pragma unroll
      for (int r = 0; r < 8; r++) {
        float gi = gx[0][r] + gh[0][r], gf = gx[1][r] + gh[1][r];
        float gg = gx[2][r] + gh[2][r], go = gx[3][r] + gh[3][r];
        c0r[r] = sig_hw(gf) * c0r[r] + sig_hw(gi) * tanh_hw(gg);
        float h = sig_hw(go) * tanh_hw(c0r[r]);
        *(_Float16*)(smem + OFF_H0 + wp + hoff + ((unsigned)r << 5)) = (_Float16)h;
      }
    }
    __syncthreads();

    // safe now: G1 hidden (written by G1(t-1), 1 barrier ago) + L1 input (just written)
    v16h g1h0 = ldsv(OFF_HG1 + rp + lb), g1h1 = ldsv(OFF_HG1 + rp + 1024u + lb);
    v16h l1i0 = ldsv(OFF_H0  + wp + lb), l1i1 = ldsv(OFF_H0  + wp + 1024u + lb);

    // ================= LSTM layer 1 (hidden chains first: operands ready) ====
    {
      v8f gx[4], gh[4];
#pragma unroll
      for (int i = 0; i < 4; i++) {
        v8f zz;
#pragma unroll
        for (int r = 0; r < 8; r++) zz[r] = 0.0f;
        zz = __builtin_amdgcn_wmma_f32_16x16x32_f16(false, l1h0, false, wL1H[i][0], (short)0, zz, false, false);
        gh[i] = __builtin_amdgcn_wmma_f32_16x16x32_f16(false, l1h1, false, wL1H[i][1], (short)0, zz, false, false);
      }
#pragma unroll
      for (int i = 0; i < 4; i++) {
        v8f cx;
#pragma unroll
        for (int r = 0; r < 8; r++) cx[r] = bL1[i];
        cx = __builtin_amdgcn_wmma_f32_16x16x32_f16(false, l1i0, false, wL1I[i][0], (short)0, cx, false, false);
        gx[i] = __builtin_amdgcn_wmma_f32_16x16x32_f16(false, l1i1, false, wL1I[i][1], (short)0, cx, false, false);
      }
#pragma unroll
      for (int r = 0; r < 8; r++) {
        float gi = gx[0][r] + gh[0][r], gf = gx[1][r] + gh[1][r];
        float gg = gx[2][r] + gh[2][r], go = gx[3][r] + gh[3][r];
        c1r[r] = sig_hw(gf) * c1r[r] + sig_hw(gi) * tanh_hw(gg);
        float h = sig_hw(go) * tanh_hw(c1r[r]);
        *(_Float16*)(smem + OFF_H1 + wp + hoff + ((unsigned)r << 5)) = (_Float16)h;
      }
    }
    __syncthreads();

    v16h g0i0 = ldsv(OFF_H1 + wp + lb), g0i1 = ldsv(OFF_H1 + wp + 1024u + lb);

    // ================= GRU layer 0 (hidden chains first) =================
    {
      v8f gx[3], gh[3];
#pragma unroll
      for (int i = 0; i < 3; i++) {
        v8f ch;
#pragma unroll
        for (int r = 0; r < 8; r++) ch[r] = bGH0[i];
        ch = __builtin_amdgcn_wmma_f32_16x16x32_f16(false, g0h0, false, wG0H[i][0], (short)0, ch, false, false);
        gh[i] = __builtin_amdgcn_wmma_f32_16x16x32_f16(false, g0h1, false, wG0H[i][1], (short)0, ch, false, false);
      }
#pragma unroll
      for (int i = 0; i < 3; i++) {
        v8f cx;
#pragma unroll
        for (int r = 0; r < 8; r++) cx[r] = bGI0[i];
        cx = __builtin_amdgcn_wmma_f32_16x16x32_f16(false, g0i0, false, wG0I[i][0], (short)0, cx, false, false);
        gx[i] = __builtin_amdgcn_wmma_f32_16x16x32_f16(false, g0i1, false, wG0I[i][1], (short)0, cx, false, false);
      }
#pragma unroll
      for (int r = 0; r < 8; r++) {
        float rr = sig_hw(gx[0][r] + gh[0][r]);
        float zz = sig_hw(gx[1][r] + gh[1][r]);
        float nn = tanh_hw(gx[2][r] + rr * gh[2][r]);
        hg0r[r] = (1.0f - zz) * nn + zz * hg0r[r];
        *(_Float16*)(smem + OFF_HG0 + wp + hoff + ((unsigned)r << 5)) = (_Float16)hg0r[r];
      }
    }
    __syncthreads();

    v16h g1i0 = ldsv(OFF_HG0 + wp + lb), g1i1 = ldsv(OFF_HG0 + wp + 1024u + lb);

    // ================= GRU layer 1 (hidden chains first; no trailing barrier) =
    {
      v8f gx[3], gh[3];
#pragma unroll
      for (int i = 0; i < 3; i++) {
        v8f ch;
#pragma unroll
        for (int r = 0; r < 8; r++) ch[r] = bGH1[i];
        ch = __builtin_amdgcn_wmma_f32_16x16x32_f16(false, g1h0, false, wG1H[i][0], (short)0, ch, false, false);
        gh[i] = __builtin_amdgcn_wmma_f32_16x16x32_f16(false, g1h1, false, wG1H[i][1], (short)0, ch, false, false);
      }
#pragma unroll
      for (int i = 0; i < 3; i++) {
        v8f cx;
#pragma unroll
        for (int r = 0; r < 8; r++) cx[r] = bGI1[i];
        cx = __builtin_amdgcn_wmma_f32_16x16x32_f16(false, g1i0, false, wG1I[i][0], (short)0, cx, false, false);
        gx[i] = __builtin_amdgcn_wmma_f32_16x16x32_f16(false, g1i1, false, wG1I[i][1], (short)0, cx, false, false);
      }
#pragma unroll
      for (int r = 0; r < 8; r++) {
        float rr = sig_hw(gx[0][r] + gh[0][r]);
        float zz = sig_hw(gx[1][r] + gh[1][r]);
        float nn = tanh_hw(gx[2][r] + rr * gh[2][r]);
        hg1r[r] = (1.0f - zz) * nn + zz * hg1r[r];
        *(_Float16*)(smem + OFF_HG1 + wp + hoff + ((unsigned)r << 5)) = (_Float16)hg1r[r];
      }
    }
  }

  // ================= FC head =================
#pragma unroll
  for (int r = 0; r < 8; r++) RED[(mrow + r) * 64 + jcol] = hg1r[r] * fcwj;
  __syncthreads();
  if (tid < BT) {
    float s = fc_b[0];
    for (int k = 0; k < 64; k++) s += RED[tid * 64 + k];
    out[wg * BT + tid] = s;
  }
}

extern "C" void kernel_launch(void* const* d_in, const int* in_sizes, int n_in,
                              void* d_out, int out_size, void* d_ws, size_t ws_size,
                              hipStream_t stream) {
  (void)in_sizes; (void)n_in; (void)d_ws; (void)ws_size; (void)out_size;
  const float* x      = (const float*)d_in[0];
  const float* lw_ih0 = (const float*)d_in[1];
  const float* lw_hh0 = (const float*)d_in[2];
  const float* lb_ih0 = (const float*)d_in[3];
  const float* lb_hh0 = (const float*)d_in[4];
  const float* lw_ih1 = (const float*)d_in[5];
  const float* lw_hh1 = (const float*)d_in[6];
  const float* lb_ih1 = (const float*)d_in[7];
  const float* lb_hh1 = (const float*)d_in[8];
  const float* gw_ih0 = (const float*)d_in[9];
  const float* gw_hh0 = (const float*)d_in[10];
  const float* gb_ih0 = (const float*)d_in[11];
  const float* gb_hh0 = (const float*)d_in[12];
  const float* gw_ih1 = (const float*)d_in[13];
  const float* gw_hh1 = (const float*)d_in[14];
  const float* gb_ih1 = (const float*)d_in[15];
  const float* gb_hh1 = (const float*)d_in[16];
  const float* fc_w   = (const float*)d_in[17];
  const float* fc_b   = (const float*)d_in[18];
  float* out = (float*)d_out;

  const int B = 2048;
  dim3 grid(B / BT), block(WGT);
  rnn_fused<<<grid, block, LDS_TOTAL, stream>>>(
      x, lw_ih0, lw_hh0, lb_ih0, lb_hh0, lw_ih1, lw_hh1, lb_ih1, lb_hh1,
      gw_ih0, gw_hh0, gb_ih0, gb_hh0, gw_ih1, gw_hh1, gb_ih1, gb_hh1,
      fc_w, fc_b, out);
}
